// StreamAwareLoRA_26018911879622
// MI455X (gfx1250) — matrix operand order repaired and checked by reading, not compiled, not run on hardware
//
#include <hip/hip_runtime.h>

typedef __attribute__((ext_vector_type(16))) _Float16 v16h;
typedef __attribute__((ext_vector_type(8)))  _Float16 v8h;
typedef __attribute__((ext_vector_type(8)))  float    v8f;

// Problem constants (match reference)
#define D_IN   4096
#define D_OUT  4096
#define MTOT   65536            // 32 * 2048 rows
#define RANK   16
#define SCALING 2.0f            // 32.0 / 16
#define STREAM_ROWS 8192        // (32/8) * 2048 rows per LoRA stream

// Tiling
#define BM 128
#define BN 128
#define BK 32
#define KSTEPS (D_IN / BK)      // 128
#define LDSS (BK + 8)           // padded f16 row stride (80 B) to dodge bank conflicts

union F16x16 { v16h v; v8h h[2]; };

__device__ __forceinline__ v8h cvt8(float4 a, float4 b) {
    v8h r;
    r[0] = (_Float16)a.x; r[1] = (_Float16)a.y; r[2] = (_Float16)a.z; r[3] = (_Float16)a.w;
    r[4] = (_Float16)b.x; r[5] = (_Float16)b.y; r[6] = (_Float16)b.z; r[7] = (_Float16)b.w;
    return r;
}

__device__ __forceinline__ v8h zero8() {
    v8h z;
    #pragma unroll
    for (int i = 0; i < 8; ++i) z[i] = (_Float16)0.0f;
    return z;
}

// Load a 128-row x 32-col fp32 tile -> f16 LDS tile. 256 threads, 2 per row.
__device__ __forceinline__ void load_tile128(const float* __restrict__ src, long srcStride,
                                             _Float16* dst, int t) {
    const int row  = t >> 1;
    const int kofs = (t & 1) << 4;          // 0 or 16
    const float4* p = (const float4*)(src + (long)row * srcStride + kofs);
    float4 f0 = p[0], f1 = p[1], f2 = p[2], f3 = p[3];
    v8h* d = (v8h*)(dst + row * LDSS + kofs);
    d[0] = cvt8(f0, f1);
    d[1] = cvt8(f2, f3);
}

// ---------------------------------------------------------------------------
// Kernel 1: aScaled[m, r] = SCALING * sum_k x[m,k] * lora_A[p][r][k]  (f16 out)
// One block = 128 rows x 16 ranks. 8 waves, each wave one 16x16 WMMA tile.
// ---------------------------------------------------------------------------
__global__ void __launch_bounds__(256)
lora_a_kernel(const float* __restrict__ x, const float* __restrict__ lA,
              _Float16* __restrict__ aOut) {
    __shared__ __align__(16) _Float16 As[2][BM][LDSS];
    __shared__ __align__(16) _Float16 Bs[2][RANK][LDSS];

    const int t    = threadIdx.x;
    const int lane = t & 31;
    const int w    = t >> 5;                // 0..7, wave's 16-row slice
    const int lr   = lane & 15;
    const int hi   = lane >> 4;
    const int aks  = hi << 3;               // A frag K chunk: 0 / 8
    const int bks  = hi << 3;               // B frag K chunk: 0 / 8

    const long m0 = (long)blockIdx.x * BM;
    const int  p  = (int)(m0 / STREAM_ROWS);
    const float* Ap = lA + ((long)p * RANK) * D_IN;

    v8f acc;
    #pragma unroll
    for (int i = 0; i < 8; ++i) acc[i] = 0.0f;

    // k-step loader
    auto load = [&](int buf, int kb) {
        const long k = (long)kb * BK;
        load_tile128(x + m0 * D_IN + k, D_IN, &As[buf][0][0], t);
        if (t < 64) {
            const int r    = t >> 2;
            const int kofs = (t & 3) << 3;  // 0,8,16,24
            const float4* s = (const float4*)(Ap + (long)r * D_IN + k + kofs);
            *(v8h*)(&Bs[buf][r][kofs]) = cvt8(s[0], s[1]);
        }
    };

    load(0, 0);
    __syncthreads();

    for (int kb = 0; kb < KSTEPS; ++kb) {
        const int cur = kb & 1;
        if (kb + 1 < KSTEPS) load(cur ^ 1, kb + 1);

        F16x16 a, b;
        const _Float16* ar = &As[cur][(w << 4) + lr][0];
        a.h[0] = *(const v8h*)(ar + aks);
        a.h[1] = *(const v8h*)(ar + aks + 16);
        const _Float16* br = &Bs[cur][lr][0];
        b.h[0] = *(const v8h*)(br + bks);
        b.h[1] = *(const v8h*)(br + bks + 16);

        acc = __builtin_amdgcn_wmma_f32_16x16x32_f16(false, a.v, false, b.v,
                                                     (short)0, acc, false, false);
        if (kb + 1 < KSTEPS) __syncthreads();
    }

    // Store scaled f16 result: lane lr = rank, rows w*16 + hi*8 + v
    const long mb = m0 + (w << 4) + (hi << 3);
    #pragma unroll
    for (int v = 0; v < 8; ++v)
        aOut[(mb + v) * RANK + lr] = (_Float16)(acc[v] * SCALING);
}

// ---------------------------------------------------------------------------
// Kernel 2: out = x @ W^T + b  (+ rank-16 LoRA folded in as one extra K-step)
// 128x128 tile, BK=32, 8 waves in 2(M) x 4(N), wave tile 64x32 (4x2 WMMA).
// ---------------------------------------------------------------------------
__global__ void __launch_bounds__(256)
gemm_kernel(const float* __restrict__ x, const float* __restrict__ W,
            const float* __restrict__ bias, const float* __restrict__ lB,
            const _Float16* __restrict__ aScaled, float* __restrict__ out) {
    __shared__ __align__(16) _Float16 As[2][BM][LDSS];
    __shared__ __align__(16) _Float16 Bs[2][BN][LDSS];

    // Grouped-M swizzle: reuse an 8-block x-stripe (16 MB) from L2 across all N.
    const int GRID_M = MTOT / BM;   // 512
    const int GRID_N = D_OUT / BN;  // 32
    const int GROUP  = 8;
    int bid   = blockIdx.x;
    int group = bid / (GROUP * GRID_N);
    int rem   = bid % (GROUP * GRID_N);
    int gm    = group * GROUP;
    int msz   = (GRID_M - gm) < GROUP ? (GRID_M - gm) : GROUP;
    const int bm = gm + rem % msz;
    const int bn = rem / msz;

    const long m0 = (long)bm * BM;
    const long n0 = (long)bn * BN;
    const int  p  = (int)(m0 / STREAM_ROWS);

    const int t     = threadIdx.x;
    const int lane  = t & 31;
    const int w     = t >> 5;
    const int waveM = w >> 2;               // 0..1
    const int waveN = w & 3;                // 0..3
    const int lr    = lane & 15;
    const int hi    = lane >> 4;
    const int aks   = hi << 3;
    const int bks   = hi << 3;

    v8f c[4][2];
    #pragma unroll
    for (int i = 0; i < 4; ++i)
        #pragma unroll
        for (int j = 0; j < 2; ++j)
            #pragma unroll
            for (int e = 0; e < 8; ++e) c[i][j][e] = 0.0f;

    auto load = [&](int buf, int kb) {
        const long k = (long)kb * BK;
        load_tile128(x + m0 * D_IN + k, D_IN, &As[buf][0][0], t);
        load_tile128(W + n0 * D_IN + k, D_IN, &Bs[buf][0][0], t);
    };

    // Extra K-step carrying the LoRA rank-16 update (cols 16..31 zero-padded).
    auto load_lora = [&](int buf) {
        const int row  = t >> 1;
        const int half = t & 1;
        if (half == 0) {
            const v8h* s = (const v8h*)(aScaled + (m0 + row) * RANK);
            v8h* d = (v8h*)(&As[buf][row][0]);
            d[0] = s[0]; d[1] = s[1];
            const float4* q = (const float4*)(lB + ((long)p * D_OUT + (n0 + row)) * RANK);
            v8h* e = (v8h*)(&Bs[buf][row][0]);
            e[0] = cvt8(q[0], q[1]);
            e[1] = cvt8(q[2], q[3]);
        } else {
            v8h z = zero8();
            v8h* d = (v8h*)(&As[buf][row][16]);
            d[0] = z; d[1] = z;
            v8h* e = (v8h*)(&Bs[buf][row][16]);
            e[0] = z; e[1] = z;
        }
    };

    load(0, 0);
    __syncthreads();

    for (int kb = 0; kb <= KSTEPS; ++kb) {      // kb == KSTEPS is the LoRA step
        const int cur = kb & 1;
        if (kb < KSTEPS) {
            if (kb + 1 < KSTEPS) load(cur ^ 1, kb + 1);
            else                 load_lora(cur ^ 1);
        }

        F16x16 a[4], b[2];
        #pragma unroll
        for (int i = 0; i < 4; ++i) {
            const _Float16* r = &As[cur][(waveM << 6) + (i << 4) + lr][0];
            a[i].h[0] = *(const v8h*)(r + aks);
            a[i].h[1] = *(const v8h*)(r + aks + 16);
        }
        #pragma unroll
        for (int j = 0; j < 2; ++j) {
            const _Float16* r = &Bs[cur][(waveN << 5) + (j << 4) + lr][0];
            b[j].h[0] = *(const v8h*)(r + bks);
            b[j].h[1] = *(const v8h*)(r + bks + 16);
        }
        #pragma unroll
        for (int i = 0; i < 4; ++i)
            #pragma unroll
            for (int j = 0; j < 2; ++j)
                c[i][j] = __builtin_amdgcn_wmma_f32_16x16x32_f16(
                    false, a[i].v, false, b[j].v, (short)0, c[i][j], false, false);

        if (kb < KSTEPS) __syncthreads();
    }

    // Epilogue: add bias, store fp32.
    #pragma unroll
    for (int j = 0; j < 2; ++j) {
        const long col = n0 + (waveN << 5) + (j << 4) + lr;
        const float bj = bias[col];
        #pragma unroll
        for (int i = 0; i < 4; ++i) {
            const long rbase = m0 + (waveM << 6) + (i << 4) + (hi << 3);
            #pragma unroll
            for (int v = 0; v < 8; ++v)
                out[(rbase + v) * (long)D_OUT + col] = c[i][j][v] + bj;
        }
    }
}

extern "C" void kernel_launch(void* const* d_in, const int* in_sizes, int n_in,
                              void* d_out, int out_size, void* d_ws, size_t ws_size,
                              hipStream_t stream) {
    const float* x    = (const float*)d_in[0];   // (32, 2048, 4096)
    const float* W    = (const float*)d_in[1];   // (4096, 4096)
    const float* bias = (const float*)d_in[2];   // (4096,)
    const float* lA   = (const float*)d_in[3];   // (8, 16, 4096)
    const float* lB   = (const float*)d_in[4];   // (8, 4096, 16)
    float* out        = (float*)d_out;
    _Float16* aTmp    = (_Float16*)d_ws;         // 65536*16 f16 = 2 MB

    lora_a_kernel<<<MTOT / BM, 256, 0, stream>>>(x, lA, aTmp);
    gemm_kernel<<<(MTOT / BM) * (D_OUT / BN), 256, 0, stream>>>(x, W, bias, lB, aTmp, out);
}
